// RotarySelfAttention_48258252538367
// MI455X (gfx1250) — matrix operand order repaired and checked by reading, not compiled, not run on hardware
//
#include <hip/hip_runtime.h>
#include <stdint.h>

#define B_  2
#define S_  2048
#define D_  1024
#define H_  16
#define DH_ 64

#define GAS __attribute__((address_space(1)))
#define LAS __attribute__((address_space(3)))

typedef __attribute__((ext_vector_type(16))) __bf16        v16bf;
typedef __attribute__((ext_vector_type(8)))  float         v8f;
typedef __attribute__((ext_vector_type(8)))  unsigned int  v8u;
typedef __attribute__((ext_vector_type(4)))  unsigned int  u32x4;
typedef __attribute__((ext_vector_type(8)))  int           i32x8;
typedef __attribute__((ext_vector_type(4)))  int           i32x4;

#if __has_builtin(__builtin_amdgcn_tensor_load_to_lds) && __has_builtin(__builtin_amdgcn_s_wait_tensorcnt)
#define USE_TDM 1
#endif
#if __has_builtin(__builtin_amdgcn_global_load_async_to_lds_b128) && __has_builtin(__builtin_amdgcn_s_wait_asynccnt)
#define USE_ASYNC 1
#endif

__device__ __forceinline__ unsigned short f2bf(float f) {
  unsigned int u = __float_as_uint(f);
  u += 0x7FFFu + ((u >> 16) & 1u);          // round-to-nearest-even
  return (unsigned short)(u >> 16);
}
__device__ __forceinline__ float bf2f(unsigned short h) {
  return __uint_as_float(((unsigned int)h) << 16);
}

// ---- WMMA fragment loaders (CDNA5 ISA 7.12.2 layouts) -----------------------
__device__ __forceinline__ v16bf load_frag_a(const unsigned short* t, int ld, int lane) {
  const int m = lane & 15, half = lane >> 4;
  const uint4 a = *(const uint4*)(t + m * ld + 8 * half);
  const uint4 b = *(const uint4*)(t + m * ld + 16 + 8 * half);
  v8u u; u[0]=a.x; u[1]=a.y; u[2]=a.z; u[3]=a.w; u[4]=b.x; u[5]=b.y; u[6]=b.z; u[7]=b.w;
  return __builtin_bit_cast(v16bf, u);
}
__device__ __forceinline__ v16bf load_frag_b(const unsigned short* t, int ld, int lane) {
  const int n = lane & 15, half = lane >> 4;
  const uint4 a = *(const uint4*)(t + n * ld + 8 * half);
  const uint4 b = *(const uint4*)(t + n * ld + 16 + 8 * half);
  v8u u; u[0]=a.x; u[1]=a.y; u[2]=a.z; u[3]=a.w; u[4]=b.x; u[5]=b.y; u[6]=b.z; u[7]=b.w;
  return __builtin_bit_cast(v16bf, u);
}
__device__ __forceinline__ v8f wmma_bf16(v16bf a, v16bf b, v8f c) {
  return __builtin_amdgcn_wmma_f32_16x16x32_bf16(false, a, false, b, (short)0, c, false, false);
}

#ifdef USE_TDM
__device__ __forceinline__ unsigned lds_raw(const void* p) {
  return (unsigned)(size_t)(LAS const void*)p;   // addrspace(3) ptr = raw LDS byte offset
}
// 2D bf16 tile load via Tensor Data Mover.  Tile: tile_x elems/row, tile_y rows,
// row stride stride_x elems; packed contiguously into LDS at ldst.
__device__ __forceinline__ void tdm_load_2d_bf16(const void* gsrc, const void* ldst,
                                                 unsigned tile_x, unsigned tile_y,
                                                 unsigned stride_x) {
  size_t ga = (size_t)gsrc;
  u32x4 g0;
  g0[0] = 1u;                                              // count=1, user mode
  g0[1] = lds_raw(ldst);                                   // lds_addr
  g0[2] = (unsigned)ga;                                    // global_addr lo
  g0[3] = (unsigned)((ga >> 32) & 0x01FFFFFFu) | 0x80000000u;  // ga hi | type=2
  const unsigned tensor_x = 1u << 30;                      // no OOB clipping
  const unsigned tensor_y = tile_y;
  i32x8 g1;
  g1[0] = (int)(1u << 16);                                 // data_size = 2 bytes
  g1[1] = (int)((tensor_x & 0xFFFFu) << 16);               // tensor_dim0 lo16
  g1[2] = (int)((tensor_x >> 16) | ((tensor_y & 0xFFFFu) << 16));
  g1[3] = (int)((tensor_y >> 16) | (tile_x << 16));        // tile_dim0
  g1[4] = (int)tile_y;                                     // tile_dim1 (tile_dim2=0)
  g1[5] = (int)stride_x;                                   // tensor_dim0_stride lo32
  g1[6] = 0;
  g1[7] = 0;
  i32x4 z4 = {0, 0, 0, 0};
  i32x8 z8 = {0, 0, 0, 0, 0, 0, 0, 0};
  __builtin_amdgcn_tensor_load_to_lds(g0, g1, z4, z4, z8, 0);
}
#endif

// ---- f32 -> bf16 convert ----------------------------------------------------
__global__ void cvt_f32_bf16(const float* __restrict__ s, unsigned short* __restrict__ d, int n) {
  int i = blockIdx.x * blockDim.x + threadIdx.x;
  int stride = gridDim.x * blockDim.x;
  for (; i < n; i += stride) d[i] = f2bf(s[i]);
}

// ---- Tiled WMMA GEMM:  C[M,N] = A[M,K] * W[N,K]^T + bias ---------------------
template <bool BF16OUT>
__global__ __launch_bounds__(128) void gemm_kernel(const unsigned short* __restrict__ A,
                                                   const unsigned short* __restrict__ W,
                                                   const float* __restrict__ bias,
                                                   float* __restrict__ Cf,
                                                   unsigned short* __restrict__ Cb,
                                                   int M, int N, int K) {
  __shared__ __align__(16) unsigned short As[64 * 32];
  __shared__ __align__(16) unsigned short Bs[64 * 32];
  const int tid  = threadIdx.x;
  const int lane = tid & 31;
  const int wave = tid >> 5;
  const int wr = wave >> 1, wc = wave & 1;
  const int row0 = blockIdx.x * 64;
  const int col0 = blockIdx.y * 64;

  v8f acc[2][2] = {};

  for (int k0 = 0; k0 < K; k0 += 32) {
#ifdef USE_TDM
    if (wave == 0) {                                 // one TDM descriptor per tile
      tdm_load_2d_bf16(A + (size_t)row0 * K + k0, As, 32, 64, (unsigned)K);
      tdm_load_2d_bf16(W + (size_t)col0 * K + k0, Bs, 32, 64, (unsigned)K);
      __builtin_amdgcn_s_wait_tensorcnt(0);
    }
#else
    for (int j = 0; j < 2; ++j) {                    // stage 64x32 A and B tiles
      int idx = tid + j * 128;                       // 0..255 (16B chunks)
      int r   = idx >> 2;
      int cg  = (idx & 3) * 8;
      *(uint4*)(As + r * 32 + cg) = *(const uint4*)(A + (size_t)(row0 + r) * K + k0 + cg);
      *(uint4*)(Bs + r * 32 + cg) = *(const uint4*)(W + (size_t)(col0 + r) * K + k0 + cg);
    }
#endif
    if (k0 + 32 < K) {                               // global_prefetch_b8 of next tiles
      __builtin_prefetch(A + (size_t)(row0 + (tid >> 1)) * K + k0 + 32, 0, 1);
      __builtin_prefetch(W + (size_t)(col0 + (tid >> 1)) * K + k0 + 32, 0, 1);
    }
    __syncthreads();
    v16bf a0 = load_frag_a(As + (wr * 32 + 0)  * 32, 32, lane);
    v16bf a1 = load_frag_a(As + (wr * 32 + 16) * 32, 32, lane);
    v16bf b0 = load_frag_b(Bs + (wc * 32 + 0)  * 32, 32, lane);
    v16bf b1 = load_frag_b(Bs + (wc * 32 + 16) * 32, 32, lane);
    acc[0][0] = wmma_bf16(a0, b0, acc[0][0]);
    acc[0][1] = wmma_bf16(a0, b1, acc[0][1]);
    acc[1][0] = wmma_bf16(a1, b0, acc[1][0]);
    acc[1][1] = wmma_bf16(a1, b1, acc[1][1]);
    __syncthreads();
  }

  const int half = lane >> 4, n15 = lane & 15;
  for (int mi = 0; mi < 2; ++mi)
    for (int ni = 0; ni < 2; ++ni) {
      int col = col0 + wc * 32 + ni * 16 + n15;
      float bv = bias[col];
      for (int r = 0; r < 8; ++r) {
        int row = row0 + wr * 32 + mi * 16 + r + 8 * half;
        float val = acc[mi][ni][r] + bv;
        if constexpr (BF16OUT) Cb[(size_t)row * N + col] = f2bf(val);
        else                   Cf[(size_t)row * N + col] = val;
      }
    }
}

// ---- RoPE + KV-cache splice + new_lengths (elementwise over [B,H,S,DH]) -----
__global__ void rope_splice_kernel(const unsigned short* __restrict__ qraw,
                                   const unsigned short* __restrict__ kraw,
                                   const unsigned short* __restrict__ vraw,
                                   const float* __restrict__ cosT,
                                   const float* __restrict__ sinT,
                                   const float* __restrict__ past_k,
                                   const float* __restrict__ past_v,
                                   const int* __restrict__ past_len,
                                   const int* __restrict__ cur_len,
                                   unsigned short* __restrict__ qb,
                                   unsigned short* __restrict__ kb,
                                   unsigned short* __restrict__ vb,
                                   float* __restrict__ kf,
                                   float* __restrict__ vf,
                                   int* __restrict__ new_len) {
  int i = blockIdx.x * blockDim.x + threadIdx.x;
  const int total = B_ * H_ * S_ * DH_;
  if (i >= total) return;
  int dh = i & 63;
  int s  = (i >> 6) & (S_ - 1);
  int h  = (i >> 17) & (H_ - 1);
  int b  = i >> 21;
  size_t ri  = (size_t)(b * S_ + s) * D_ + h * DH_ + dh;
  int dh2    = (dh < 32) ? dh + 32 : dh - 32;
  size_t ri2 = (size_t)(b * S_ + s) * D_ + h * DH_ + dh2;
  float c  = cosT[s * DH_ + dh], sn = sinT[s * DH_ + dh];

  float qv = bf2f(qraw[ri]);
  float qr = bf2f(qraw[ri2]); qr = (dh < 32) ? -qr : qr;
  qb[i] = f2bf(qv * c + qr * sn);

  float kv = bf2f(kraw[ri]);
  float kr = bf2f(kraw[ri2]); kr = (dh < 32) ? -kr : kr;
  float kk = kv * c + kr * sn;
  int cl = past_len[b]; cl = cl < 0 ? 0 : (cl > S_ ? S_ : cl);
  if (s < cl) kk = past_k[i];
  kf[i] = kk; kb[i] = f2bf(kk);

  float vv = bf2f(vraw[ri]);
  if (s < cl) vv = past_v[i];
  vf[i] = vv; vb[i] = f2bf(vv);

  if (i < B_) {
    int nl = cur_len[i]; if (nl > S_) nl = S_;
    new_len[i] = nl;
  }
}

// ---- Flash attention: grid (B*H, S/64); 4 waves, 16 q-rows per wave ---------
__global__ __launch_bounds__(128) void attn_kernel(const unsigned short* __restrict__ qb,
                                                   const unsigned short* __restrict__ kb,
                                                   const unsigned short* __restrict__ vb,
                                                   const unsigned char* __restrict__ mask,
                                                   unsigned short* __restrict__ ctx) {
  __shared__ __align__(16) unsigned short kt[32 * 64];      // [key][d]
  __shared__ __align__(16) unsigned short vt[64 * 32];      // [d][key] (transposed)
  __shared__ __align__(16) unsigned short pb[4 * 16 * 32];  // per-wave P tiles
  const int tid = threadIdx.x, lane = tid & 31, w = tid >> 5;
  const int half = lane >> 4, n15 = lane & 15;
  const int bh = blockIdx.x;
  const int b = bh / H_, h = bh % H_;
  const int q0 = blockIdx.y * 64 + w * 16;

  const unsigned short* qp = qb + ((size_t)bh * S_ + q0) * DH_;
  const unsigned short* kp = kb + (size_t)bh * S_ * DH_;
  const unsigned short* vp = vb + (size_t)bh * S_ * DH_;

  v16bf aq0 = load_frag_a(qp, DH_, lane);        // d 0..31
  v16bf aq1 = load_frag_a(qp + 32, DH_, lane);   // d 32..63

  float Mrow[8], Lrow[8];
  v8f o[4] = {};
  for (int r = 0; r < 8; ++r) { Mrow[r] = -3.0e38f; Lrow[r] = 0.f; }
  unsigned short* pw = pb + w * (16 * 32);

  for (int kbase = 0; kbase < S_; kbase += 32) {
#ifdef USE_ASYNC
    for (int j = 0; j < 2; ++j) {                // async LDS fill of K tile
      int idx = tid + j * 128;                   // 16B chunks
      int key = idx >> 3, cg = (idx & 7) * 8;
      __builtin_amdgcn_global_load_async_to_lds_b128(
          (GAS i32x4*)(GAS void*)(void*)(kp + (size_t)(kbase + key) * DH_ + cg),
          (LAS i32x4*)(LAS void*)(void*)(kt + key * 64 + cg), 0, 0);
    }
#else
    for (int j = 0; j < 2; ++j) {                // stage 32 keys x 64 dims of K
      int idx = tid + j * 128;
      int key = idx >> 3, cg = (idx & 7) * 8;
      *(uint4*)(kt + key * 64 + cg) = *(const uint4*)(kp + (size_t)(kbase + key) * DH_ + cg);
    }
#endif
    for (int j = 0; j < 16; ++j) {               // stage V transposed [d][key]
      int idx = tid + j * 128;
      int d = idx & 63, key = idx >> 6;
      vt[d * 32 + key] = vp[(size_t)(kbase + key) * DH_ + d];
    }
#ifdef USE_ASYNC
    __builtin_amdgcn_s_wait_asynccnt(0);
#endif
    __syncthreads();

    v8f z = {};
    v8f sc0 = wmma_bf16(aq0, load_frag_b(kt, 64, lane), z);
    sc0     = wmma_bf16(aq1, load_frag_b(kt + 32, 64, lane), sc0);
    v8f sc1 = wmma_bf16(aq0, load_frag_b(kt + 16 * 64, 64, lane), z);
    sc1     = wmma_bf16(aq1, load_frag_b(kt + 16 * 64 + 32, 64, lane), sc1);

    const float scale = 0.125f;                  // 1/sqrt(64)
    const bool m0 = mask[b * S_ + kbase + n15] != 0;
    const bool m1 = mask[b * S_ + kbase + 16 + n15] != 0;
    float alpha[8];
    for (int r = 0; r < 8; ++r) {
      float s0 = m0 ? sc0[r] * scale : -3.0e38f;
      float s1 = m1 ? sc1[r] * scale : -3.0e38f;
      float mx = fmaxf(s0, s1);
      mx = fmaxf(mx, __shfl_xor(mx, 1, 32));
      mx = fmaxf(mx, __shfl_xor(mx, 2, 32));
      mx = fmaxf(mx, __shfl_xor(mx, 4, 32));
      mx = fmaxf(mx, __shfl_xor(mx, 8, 32));     // row max over the 16-lane half
      float newM = fmaxf(Mrow[r], mx);
      alpha[r] = __expf(Mrow[r] - newM);
      Mrow[r] = newM;
      float e0 = m0 ? __expf(s0 - newM) : 0.f;
      float e1 = m1 ? __expf(s1 - newM) : 0.f;
      float rs = e0 + e1;
      rs += __shfl_xor(rs, 1, 32);
      rs += __shfl_xor(rs, 2, 32);
      rs += __shfl_xor(rs, 4, 32);
      rs += __shfl_xor(rs, 8, 32);
      Lrow[r] = Lrow[r] * alpha[r] + rs;
      pw[(r + 8 * half) * 32 + n15]      = f2bf(e0);   // C-layout -> A-layout via LDS
      pw[(r + 8 * half) * 32 + 16 + n15] = f2bf(e1);
    }
    for (int nt = 0; nt < 4; ++nt)
      for (int r = 0; r < 8; ++r) o[nt][r] *= alpha[r];

    v16bf pa = load_frag_a(pw, 32, lane);        // P is 16x32 = one A fragment
    o[0] = wmma_bf16(pa, load_frag_b(vt,           32, lane), o[0]);
    o[1] = wmma_bf16(pa, load_frag_b(vt + 16 * 32, 32, lane), o[1]);
    o[2] = wmma_bf16(pa, load_frag_b(vt + 32 * 32, 32, lane), o[2]);
    o[3] = wmma_bf16(pa, load_frag_b(vt + 48 * 32, 32, lane), o[3]);
    __syncthreads();
  }

  for (int nt = 0; nt < 4; ++nt) {
    int col = h * DH_ + nt * 16 + n15;
    for (int r = 0; r < 8; ++r) {
      int srow = q0 + r + 8 * half;
      float L = Lrow[r];
      float val = (L > 0.f) ? o[nt][r] / L : 0.f;
      ctx[((size_t)b * S_ + srow) * D_ + col] = f2bf(val);
    }
  }
}

extern "C" void kernel_launch(void* const* d_in, const int* in_sizes, int n_in,
                              void* d_out, int out_size, void* d_ws, size_t ws_size,
                              hipStream_t stream) {
  (void)in_sizes; (void)n_in; (void)out_size; (void)ws_size;
  const float*         x        = (const float*)d_in[0];
  const unsigned char* mask     = (const unsigned char*)d_in[1];
  const float*         cosT     = (const float*)d_in[2];
  const float*         sinT     = (const float*)d_in[3];
  const float*         past_k   = (const float*)d_in[4];
  const float*         past_v   = (const float*)d_in[5];
  const int*           past_len = (const int*)d_in[6];
  const int*           cur_len  = (const int*)d_in[7];
  const float* Wq = (const float*)d_in[8];  const float* bq = (const float*)d_in[9];
  const float* Wk = (const float*)d_in[10]; const float* bk = (const float*)d_in[11];
  const float* Wv = (const float*)d_in[12]; const float* bv = (const float*)d_in[13];
  const float* Wo = (const float*)d_in[14]; const float* bo = (const float*)d_in[15];

  const size_t nX  = (size_t)B_ * S_ * D_;
  const size_t nW  = (size_t)D_ * D_;
  const size_t nKV = (size_t)B_ * H_ * S_ * DH_;

  float* out   = (float*)d_out;
  float* k_out = out + nX;
  float* v_out = k_out + nKV;
  int*   l_out = (int*)(v_out + nKV);

  char* ws = (char*)d_ws;
  size_t off = 0;
  auto alloc = [&](size_t bytes) {
    off = (off + 255) & ~(size_t)255;
    char* p = ws + off;
    off += bytes;
    return p;
  };
  unsigned short* xb   = (unsigned short*)alloc(nX * 2);
  unsigned short* wqb  = (unsigned short*)alloc(nW * 2);
  unsigned short* wkb  = (unsigned short*)alloc(nW * 2);
  unsigned short* wvb  = (unsigned short*)alloc(nW * 2);
  unsigned short* wob  = (unsigned short*)alloc(nW * 2);
  unsigned short* qraw = (unsigned short*)alloc(nX * 2);
  unsigned short* kraw = (unsigned short*)alloc(nX * 2);
  unsigned short* vraw = (unsigned short*)alloc(nX * 2);
  unsigned short* qbh  = (unsigned short*)alloc(nKV * 2);
  unsigned short* kbh  = (unsigned short*)alloc(nKV * 2);
  unsigned short* vbh  = (unsigned short*)alloc(nKV * 2);
  unsigned short* ctxb = (unsigned short*)alloc(nX * 2);

  cvt_f32_bf16<<<4096, 256, 0, stream>>>(x,  xb,  (int)nX);
  cvt_f32_bf16<<<1024, 256, 0, stream>>>(Wq, wqb, (int)nW);
  cvt_f32_bf16<<<1024, 256, 0, stream>>>(Wk, wkb, (int)nW);
  cvt_f32_bf16<<<1024, 256, 0, stream>>>(Wv, wvb, (int)nW);
  cvt_f32_bf16<<<1024, 256, 0, stream>>>(Wo, wob, (int)nW);

  const int M = B_ * S_;
  dim3 gg(M / 64, D_ / 64);
  gemm_kernel<true ><<<gg, 128, 0, stream>>>(xb, wqb, bq, nullptr, qraw, M, D_, D_);
  gemm_kernel<true ><<<gg, 128, 0, stream>>>(xb, wkb, bk, nullptr, kraw, M, D_, D_);
  gemm_kernel<true ><<<gg, 128, 0, stream>>>(xb, wvb, bv, nullptr, vraw, M, D_, D_);

  const int total = B_ * H_ * S_ * DH_;
  rope_splice_kernel<<<(total + 255) / 256, 256, 0, stream>>>(
      qraw, kraw, vraw, cosT, sinT, past_k, past_v, past_len, cur_len,
      qbh, kbh, vbh, k_out, v_out, l_out);

  attn_kernel<<<dim3(B_ * H_, S_ / 64), 128, 0, stream>>>(qbh, kbh, vbh, mask, ctxb);

  gemm_kernel<false><<<gg, 128, 0, stream>>>(ctxb, wob, bo, out, nullptr, M, D_, D_);
}
